// AdaptiveSparseAttention_58901181497760
// MI455X (gfx1250) — compile-verified, matrix operand order checked by reading, not run on hardware
//
#include <hip/hip_runtime.h>
#include <hip/hip_bf16.h>

// ---------------------------------------------------------------------------
// AdaptiveSparseAttention for MI455X (gfx1250), wave32 + WMMA bf16.
//   B=4, L=1024, D=512, H=8, hd=64.  All matmuls via v_wmma_f32_16x16x32_bf16.
//   V-tile staging uses gfx1250 async global->LDS copies (ASYNCcnt).
//   GEMM keeps fragments loaded in-loop (no manual pipelining): spill-free
//   ~110 VGPR body -> latency hidden by wave-level parallelism, data L2-hot.
// ---------------------------------------------------------------------------

typedef __bf16 bf16;
typedef __attribute__((ext_vector_type(16))) __bf16 v16bf;
typedef __attribute__((ext_vector_type(8)))  __bf16 v8bf;
typedef __attribute__((ext_vector_type(8)))  float  v8f;

#define B_      4
#define L_      1024
#define D_      512
#define H_      8
#define HD_     64
#define QKV_LD  1536          // 3*D
#define KK_     716           // int(L * (1 - 0.3))
#define HALFWIN 16            // WIN/2
#define THRESH_C 0.1f
#define SCALE_C  0.125f       // hd^-0.5 = 1/8

__device__ __forceinline__ float neg_inf() { return -__builtin_inff(); }

// CDNA5 16-bit A/B fragment (16x32 or 32x16), per ISA 7.12.2 layout:
// lane = hi*16 + r ; lane holds K = hi*8+{0..7} and 16+hi*8+{0..7} of row/col r.
__device__ __forceinline__ v16bf load_frag(const bf16* __restrict__ base, int ld, int lane) {
  const int r = lane & 15, hi = lane >> 4;
  const v8bf c0 = *(const v8bf*)(base + (size_t)r * ld + hi * 8);
  const v8bf c1 = *(const v8bf*)(base + (size_t)r * ld + 16 + hi * 8);
  v16bf f;
#pragma unroll
  for (int i = 0; i < 8; ++i) { f[i] = c0[i]; f[8 + i] = c1[i]; }
  return f;
}

__device__ __forceinline__ v8f wmma_bf16(v16bf a, v16bf b, v8f c) {
  return __builtin_amdgcn_wmma_f32_16x16x32_bf16(false, a, false, b, (short)0, c,
                                                 false, false);
}

// ---------------------------------------------------------------------------
// 1) fp32 -> bf16 cast
// ---------------------------------------------------------------------------
__global__ void k_cast_bf16(const float* __restrict__ src, bf16* __restrict__ dst, int n) {
  int i = blockIdx.x * blockDim.x + threadIdx.x;
  if (i < n) dst[i] = (bf16)src[i];
}

// ---------------------------------------------------------------------------
// 2) pattern selector MLP -> pw[B][3]   (one 256-thread block per batch)
// ---------------------------------------------------------------------------
__global__ void k_pattern(const float* __restrict__ x,
                          const float* __restrict__ w1, const float* __restrict__ b1,
                          const float* __restrict__ w2, const float* __restrict__ b2,
                          const float* __restrict__ w3, const float* __restrict__ b3,
                          const float* __restrict__ pbias, float* __restrict__ pw) {
  __shared__ float pooled[D_];
  __shared__ float h1[D_];
  __shared__ float h2[D_ / 2];
  __shared__ float lg[3];
  const int b = blockIdx.x, t = threadIdx.x;

  for (int c = t; c < D_; c += 256) {
    float s = 0.f;
    for (int i = 0; i < L_; ++i) s += x[((size_t)b * L_ + i) * D_ + c];
    pooled[c] = s * (1.0f / (float)L_);
  }
  __syncthreads();
  for (int o = t; o < D_; o += 256) {
    float s = b1[o];
    for (int i = 0; i < D_; ++i) s += pooled[i] * w1[(size_t)o * D_ + i];
    h1[o] = fmaxf(s, 0.f);
  }
  __syncthreads();
  for (int o = t; o < D_ / 2; o += 256) {
    float s = b2[o];
    for (int i = 0; i < D_; ++i) s += h1[i] * w2[(size_t)o * D_ + i];
    h2[o] = fmaxf(s, 0.f);
  }
  __syncthreads();
  if (t < 3) {
    float s = b3[t] + pbias[t];
    for (int i = 0; i < D_ / 2; ++i) s += h2[i] * w3[(size_t)t * (D_ / 2) + i];
    lg[t] = s * 2.0f;                       // / PTEMP (0.5)
  }
  __syncthreads();
  if (t == 0) {
    float m = fmaxf(lg[0], fmaxf(lg[1], lg[2]));
    float e0 = __expf(lg[0] - m), e1 = __expf(lg[1] - m), e2 = __expf(lg[2] - m);
    float inv = 1.f / (e0 + e1 + e2);
    pw[b * 3 + 0] = e0 * inv; pw[b * 3 + 1] = e1 * inv; pw[b * 3 + 2] = e2 * inv;
  }
}

// ---------------------------------------------------------------------------
// 3) GEMM core: C[16 x 64] tile per wave, fragments loaded in-loop (the
//    compiler forms load clauses; latency is hidden by multi-wave occupancy).
//    A[M][K] row-major, W[N][K] row-major (B = W^T), K = 512.
// ---------------------------------------------------------------------------
template <int OUT_F32>
__device__ __forceinline__ void gemm_tile(const bf16* __restrict__ A,
                                          const bf16* __restrict__ W,
                                          const float* __restrict__ bias,
                                          bf16* __restrict__ Cb, float* __restrict__ Cf,
                                          int m0, int n0, int ldc, int lane) {
  v8f acc[4] = {};
  for (int k0 = 0; k0 < D_; k0 += 32) {
    v16bf a = load_frag(A + (size_t)m0 * D_ + k0, D_, lane);
#pragma unroll
    for (int j = 0; j < 4; ++j) {
      v16bf bw = load_frag(W + (size_t)(n0 + j * 16) * D_ + k0, D_, lane);
      acc[j] = wmma_bf16(a, bw, acc[j]);
    }
    __builtin_prefetch(A + (size_t)m0 * D_ + k0 + 32, 0, 1);
  }
  const int r = lane & 15, hi = lane >> 4;
#pragma unroll
  for (int j = 0; j < 4; ++j)
#pragma unroll
    for (int i = 0; i < 8; ++i) {
      const size_t idx = (size_t)(m0 + hi * 8 + i) * ldc + n0 + j * 16 + r;
      if (OUT_F32) Cf[idx] = acc[j][i] + bias[n0 + j * 16 + r];
      else         Cb[idx] = (bf16)acc[j][i];
    }
}

__global__ void k_gemm_qkv(const bf16* __restrict__ A, const bf16* __restrict__ W,
                           bf16* __restrict__ C) {
  const int lane = threadIdx.x & 31;
  const int ntn = QKV_LD / 64;                       // 24
  const int n0 = (blockIdx.x % ntn) * 64;
  const int m0 = (blockIdx.x / ntn) * 16;
  gemm_tile<0>(A, W, nullptr, C, nullptr, m0, n0, QKV_LD, lane);
}

__global__ void k_gemm_proj(const bf16* __restrict__ A, const bf16* __restrict__ W,
                            const float* __restrict__ bias, float* __restrict__ C) {
  const int lane = threadIdx.x & 31;
  const int ntn = D_ / 64;                           // 8
  const int n0 = (blockIdx.x % ntn) * 64;
  const int m0 = (blockIdx.x / ntn) * 16;
  gemm_tile<1>(A, W, bias, nullptr, C, m0, n0, D_, lane);
}

// ---------------------------------------------------------------------------
// 4) per-(b,h,q) top-kk threshold on s = scale*q.k*sparse_w + sparse_b
//    one wave per row; row values staged in LDS; 24-step value binary search
// ---------------------------------------------------------------------------
__global__ void k_thresh(const bf16* __restrict__ qkv,
                         const float* __restrict__ sparse_w,
                         const float* __restrict__ sparse_b,
                         float* __restrict__ thresh) {
  __shared__ float Sv[8][32][32];                    // [wave][lane][key-chunk]
  const int wslot = threadIdx.x >> 5;
  const int lane = threadIdx.x & 31;
  const int wid = (blockIdx.x * blockDim.x + threadIdx.x) >> 5;
  if (wid >= B_ * H_ * L_) return;
  const int q = wid % L_;
  const int h = (wid / L_) % H_;
  const int b = wid / (L_ * H_);

  const bf16* qrow = qkv + ((size_t)(b * L_ + q)) * QKV_LD + h * HD_;
  float qr[HD_];
#pragma unroll
  for (int d = 0; d < HD_; ++d) qr[d] = (float)qrow[d];
  const float sw = sparse_w[h], sb = sparse_b[h];

  float mn = __builtin_inff(), mx = neg_inf();
  for (int j = 0; j < 32; ++j) {
    const int key = lane + 32 * j;
    const bf16* krow = qkv + ((size_t)(b * L_ + key)) * QKV_LD + D_ + h * HD_;
    float dot = 0.f;
#pragma unroll 8
    for (int d = 0; d < HD_; ++d) dot += qr[d] * (float)krow[d];
    const float v = dot * SCALE_C * sw + sb;
    Sv[wslot][lane][j] = v;
    mn = fminf(mn, v); mx = fmaxf(mx, v);
  }
#pragma unroll
  for (int o = 16; o >= 1; o >>= 1) {
    mn = fminf(mn, __shfl_xor(mn, o, 32));
    mx = fmaxf(mx, __shfl_xor(mx, o, 32));
  }
  float lo = mn, hi = mx;
  for (int it = 0; it < 24; ++it) {
    const float mid = 0.5f * (lo + hi);
    int cnt = 0;
    for (int j = 0; j < 32; ++j) cnt += (Sv[wslot][lane][j] >= mid) ? 1 : 0;
#pragma unroll
    for (int o = 16; o >= 1; o >>= 1) cnt += __shfl_xor(cnt, o, 32);
    if (cnt >= KK_) lo = mid; else hi = mid;
  }
  if (lane == 0) thresh[wid] = lo;
}

// ---------------------------------------------------------------------------
// 5) fused flash attention: one wave per (b, h, 16-query tile)
//    V tile is staged to LDS with gfx1250 async global->LDS copies.
// ---------------------------------------------------------------------------
__global__ void k_flash(const bf16* __restrict__ qkv, const int* __restrict__ mask,
                        const float* __restrict__ pw, const float* __restrict__ thresh,
                        const float* __restrict__ sparse_w, const float* __restrict__ sparse_b,
                        bf16* __restrict__ out) {
  __shared__ bf16 Pt[16 * 32];                       // probabilities tile
  __shared__ bf16 Vt[32 * 64];                       // staged V tile (transposed read)
  const int lane = threadIdx.x & 31;
  const int qt = blockIdx.x % (L_ / 16);
  const int h  = (blockIdx.x / (L_ / 16)) % H_;
  const int b  = blockIdx.x / ((L_ / 16) * H_);
  const int q0 = qt * 16;
  const int r = lane & 15, hi = lane >> 4;

  const bf16* Qb = qkv + ((size_t)(b * L_ + q0)) * QKV_LD + h * HD_;
  const v16bf qA0 = load_frag(Qb, QKV_LD, lane);
  const v16bf qA1 = load_frag(Qb + 32, QKV_LD, lane);

  const float p0 = pw[b * 3 + 0], p1 = pw[b * 3 + 1], p2 = pw[b * 3 + 2];
  const float sw = sparse_w[h], sb = sparse_b[h];

  float m_i[8], l_i[8], th_i[8];
#pragma unroll
  for (int i = 0; i < 8; ++i) {
    m_i[i] = neg_inf(); l_i[i] = 0.f;
    th_i[i] = thresh[((size_t)(b * H_ + h)) * L_ + q0 + hi * 8 + i];
  }
  v8f O[4] = {};

  for (int kt = 0; kt < L_; kt += 32) {
    // --- async stage of V tile (32 keys x 64 dims = 128B per lane) into LDS.
    //     INST_OFFSET is added to both the LDS and global addresses, so one
    //     base pair covers the whole row with 8 x b128 async transfers.
    {
      const bf16* Vr = qkv + ((size_t)(b * L_ + kt + lane)) * QKV_LD + 2 * D_ + h * HD_;
      unsigned lds_dst = (unsigned)(size_t)(&Vt[0]) + (unsigned)(lane * 64 * sizeof(bf16));
      unsigned long long ga = (unsigned long long)(const void*)Vr;
      asm volatile(
          "global_load_async_to_lds_b128 %0, %1, off\n\t"
          "global_load_async_to_lds_b128 %0, %1, off offset:16\n\t"
          "global_load_async_to_lds_b128 %0, %1, off offset:32\n\t"
          "global_load_async_to_lds_b128 %0, %1, off offset:48\n\t"
          "global_load_async_to_lds_b128 %0, %1, off offset:64\n\t"
          "global_load_async_to_lds_b128 %0, %1, off offset:80\n\t"
          "global_load_async_to_lds_b128 %0, %1, off offset:96\n\t"
          "global_load_async_to_lds_b128 %0, %1, off offset:112"
          :: "v"(lds_dst), "v"(ga) : "memory");
    }
    // --- scores for two 16-key subtiles (K is K-major: contiguous 16B chunks)
    v8f sc[2];
#pragma unroll
    for (int sub = 0; sub < 2; ++sub) {
      const bf16* Kb = qkv + ((size_t)(b * L_ + kt + sub * 16)) * QKV_LD + D_ + h * HD_;
      v16bf kB0 = load_frag(Kb, QKV_LD, lane);
      v16bf kB1 = load_frag(Kb + 32, QKV_LD, lane);
      v8f c = {};
      c = wmma_bf16(qA0, kB0, c);
      c = wmma_bf16(qA1, kB1, c);
      sc[sub] = c;
    }
    // --- scale + combined mask: pw0*local + pw1 + pw2*sparse > 0.1, key padding
#pragma unroll
    for (int sub = 0; sub < 2; ++sub) {
      const int key = kt + sub * 16 + r;
      const int mv = mask[b * L_ + key];
#pragma unroll
      for (int i = 0; i < 8; ++i) {
        const int qi = q0 + hi * 8 + i;
        const float v = sc[sub][i] * SCALE_C;
        int dd = qi - key; dd = dd < 0 ? -dd : dd;
        const float loc = (dd <= HALFWIN) ? 1.f : 0.f;
        const float sp  = (v * sw + sb >= th_i[i]) ? 1.f : 0.f;
        const float comb = p0 * loc + p1 + p2 * sp;
        const bool allow = (comb > THRESH_C) && (mv != 0);
        sc[sub][i] = allow ? v : neg_inf();
      }
    }
    // --- online softmax update (joint max over the 32 keys)
    float pa[2][8];
#pragma unroll
    for (int i = 0; i < 8; ++i) {
      float rm = fmaxf(sc[0][i], sc[1][i]);
#pragma unroll
      for (int o = 8; o >= 1; o >>= 1) rm = fmaxf(rm, __shfl_xor(rm, o, 32));
      const float nm = fmaxf(m_i[i], rm);
      const float alpha = (m_i[i] == neg_inf()) ? 0.f : __expf(m_i[i] - nm);
      const float e0 = (sc[0][i] == neg_inf()) ? 0.f : __expf(sc[0][i] - nm);
      const float e1 = (sc[1][i] == neg_inf()) ? 0.f : __expf(sc[1][i] - nm);
      float rs = e0 + e1;
#pragma unroll
      for (int o = 8; o >= 1; o >>= 1) rs += __shfl_xor(rs, o, 32);
      l_i[i] = l_i[i] * alpha + rs;
      m_i[i] = nm;
#pragma unroll
      for (int j = 0; j < 4; ++j) O[j][i] *= alpha;
      pa[0][i] = e0; pa[1][i] = e1;
    }
    // --- P tile to LDS (C-frag layout -> row major)
#pragma unroll
    for (int i = 0; i < 8; ++i) {
      Pt[(hi * 8 + i) * 32 + r]      = (bf16)pa[0][i];
      Pt[(hi * 8 + i) * 32 + 16 + r] = (bf16)pa[1][i];
    }
    // --- wait for async V staging + LDS writes, then P @ V via WMMA
    asm volatile("s_wait_asynccnt 0x0" ::: "memory");
    __syncthreads();
    const v16bf pA = load_frag(&Pt[0], 32, lane);
#pragma unroll
    for (int j = 0; j < 4; ++j) {
      v16bf vB;
#pragma unroll
      for (int e = 0; e < 16; ++e) {
        const int kel = hi * 8 + (e & 7) + ((e & 8) ? 16 : 0);
        vB[e] = Vt[kel * 64 + j * 16 + r];
      }
      O[j] = wmma_bf16(pA, vB, O[j]);
    }
    __syncthreads();
    if (kt + 32 < L_)
      __builtin_prefetch(qkv + ((size_t)(b * L_ + kt + 32)) * QKV_LD + D_ + h * HD_, 0, 1);
  }
  // --- epilogue: normalize; fully-masked rows fall back to V[key 0]
#pragma unroll
  for (int i = 0; i < 8; ++i) {
    const float inv = (l_i[i] > 0.f) ? 1.f / l_i[i] : 0.f;
    const int qi = q0 + hi * 8 + i;
#pragma unroll
    for (int j = 0; j < 4; ++j) {
      float val = O[j][i] * inv;
      if (l_i[i] <= 0.f)
        val = (float)qkv[((size_t)(b * L_)) * QKV_LD + 2 * D_ + h * HD_ + j * 16 + r];
      out[((size_t)(b * L_ + qi)) * D_ + h * HD_ + j * 16 + r] = (bf16)val;
    }
  }
}

// ---------------------------------------------------------------------------
extern "C" void kernel_launch(void* const* d_in, const int* in_sizes, int n_in,
                              void* d_out, int out_size, void* d_ws, size_t ws_size,
                              hipStream_t stream) {
  const float* x        = (const float*)d_in[0];
  const int*   mask     = (const int*)  d_in[1];
  const float* qkv_w    = (const float*)d_in[2];
  const float* proj_w   = (const float*)d_in[3];
  const float* proj_b   = (const float*)d_in[4];
  const float* ps_w1    = (const float*)d_in[5];
  const float* ps_b1    = (const float*)d_in[6];
  const float* ps_w2    = (const float*)d_in[7];
  const float* ps_b2    = (const float*)d_in[8];
  const float* ps_w3    = (const float*)d_in[9];
  const float* ps_b3    = (const float*)d_in[10];
  const float* pbias    = (const float*)d_in[11];
  const float* sparse_w = (const float*)d_in[12];
  const float* sparse_b = (const float*)d_in[13];
  float* out = (float*)d_out;

  char* ws = (char*)d_ws;
  const size_t nX  = (size_t)B_ * L_ * D_;           // 2,097,152
  const size_t nQW = (size_t)QKV_LD * D_;            //   786,432
  const size_t nPW = (size_t)D_ * D_;                //   262,144
  const size_t nQKV = (size_t)B_ * L_ * QKV_LD;      // 6,291,456

  bf16*  x_bf    = (bf16*)(ws);                              size_t off = nX * 2;
  bf16*  qkvw_bf = (bf16*)(ws + off);                        off += nQW * 2;
  bf16*  projw_bf= (bf16*)(ws + off);                        off += nPW * 2;
  bf16*  qkv_bf  = (bf16*)(ws + off);                        off += nQKV * 2;
  bf16*  attn_bf = (bf16*)(ws + off);                        off += nX * 2;
  off = (off + 255) & ~(size_t)255;
  float* pw      = (float*)(ws + off);                       off += 256;
  float* thresh  = (float*)(ws + off);                       // B*H*L floats

  // 1) casts
  k_cast_bf16<<<(int)((nX  + 255) / 256), 256, 0, stream>>>(x, x_bf, (int)nX);
  k_cast_bf16<<<(int)((nQW + 255) / 256), 256, 0, stream>>>(qkv_w, qkvw_bf, (int)nQW);
  k_cast_bf16<<<(int)((nPW + 255) / 256), 256, 0, stream>>>(proj_w, projw_bf, (int)nPW);
  // 2) pattern weights
  k_pattern<<<B_, 256, 0, stream>>>(x, ps_w1, ps_b1, ps_w2, ps_b2, ps_w3, ps_b3,
                                    pbias, pw);
  // 3) QKV GEMM (WMMA, 16x64 tile per wave)
  k_gemm_qkv<<<(B_ * L_ / 16) * (QKV_LD / 64), 32, 0, stream>>>(x_bf, qkvw_bf, qkv_bf);
  // 4) learned-sparse thresholds
  k_thresh<<<(B_ * H_ * L_) / 8, 256, 0, stream>>>(qkv_bf, sparse_w, sparse_b, thresh);
  // 5) fused flash attention (WMMA + async V staging)
  k_flash<<<B_ * H_ * (L_ / 16), 32, 0, stream>>>(qkv_bf, mask, pw, thresh,
                                                  sparse_w, sparse_b, attn_bf);
  // 6) output projection (WMMA)
  k_gemm_proj<<<(B_ * L_ / 16) * (D_ / 64), 32, 0, stream>>>(attn_bf, projw_bf,
                                                             proj_b, out);
}
